// ConvLinearSelfAttention_43576738185968
// MI455X (gfx1250) — matrix operand order repaired and checked by reading, not compiled, not run on hardware
//
#include <hip/hip_runtime.h>

#define BB  16
#define CC  256
#define NN  4096
#define HH  8
#define DD  64
#define HID 512
#define NSL 1024  // N-slice for the context kernel (4 slices)

typedef __bf16 bf16;
typedef __attribute__((ext_vector_type(16))) __bf16 v16bf;
typedef __attribute__((ext_vector_type(8)))  __bf16 v8bf;
typedef __attribute__((ext_vector_type(8)))  float  v8f;
typedef __attribute__((ext_vector_type(4)))  float  v4f;

__device__ __forceinline__ v8f wmma_bf16(v16bf a, v16bf b, v8f c) {
  // D = A(16x32 bf16) * B(32x16 bf16) + C(16x16 f32)
  return __builtin_amdgcn_wmma_f32_16x16x32_bf16(false, a, false, b, (short)0, c,
                                                 false, false);
}

// Async DMA: 16 bytes global -> LDS, tracked by ASYNCcnt (CDNA5 async path).
__device__ __forceinline__ void async_b128(void* lds_ptr, const void* gptr) {
  unsigned lds_off = (unsigned)(unsigned long long)lds_ptr;  // low 32 bits = LDS addr
  asm volatile("global_load_async_to_lds_b128 %0, %1, off"
               :
               : "v"(lds_off), "v"(gptr)
               : "memory");
}
__device__ __forceinline__ void wait_async() {
  asm volatile("s_wait_asynccnt 0x0" ::: "memory");
}

// A operand (16x32, 16-bit): lane holds row m=lane&15; K chunks [ko,ko+8) and
// [16+ko,16+ko+8) with ko = (lane>=16)?8:0.  base stride in bf16 elements.
__device__ __forceinline__ v16bf ld_a(const bf16* base, int stride, int m, int kbase,
                                      int lane) {
  int ko = (lane & 16) ? 8 : 0;
  const bf16* p = base + m * stride + kbase + ko;
  v8bf lo = *(const v8bf*)p;
  v8bf hi = *(const v8bf*)(p + 16);
  return __builtin_shufflevector(lo, hi, 0, 1, 2, 3, 4, 5, 6, 7, 8, 9, 10, 11, 12,
                                 13, 14, 15);
}

// B operand (32x16, 16-bit): lane holds col n=lane&15; K chunks [ko,ko+8) and
// [16+ko,16+ko+8) with ko = (lane>=16)?8:0.  Stored [n][k] in LDS.
__device__ __forceinline__ v16bf ld_b(const bf16* base, int stride, int n, int kbase,
                                      int lane) {
  int ko = (lane & 16) ? 8 : 0;
  const bf16* p = base + n * stride + kbase + ko;
  v8bf lo = *(const v8bf*)p;
  v8bf hi = *(const v8bf*)(p + 16);
  return __builtin_shufflevector(lo, hi, 0, 1, 2, 3, 4, 5, 6, 7, 8, 9, 10, 11, 12,
                                 13, 14, 15);
}

// ---------------------------------------------------------------- weight prep
__global__ __launch_bounds__(256) void k_prep(const float* __restrict__ wq,
                                              const float* __restrict__ wo,
                                              bf16* __restrict__ wqb,
                                              bf16* __restrict__ wob) {
  int i = blockIdx.x * 256 + threadIdx.x;
  if (i < 3 * HID * CC) wqb[i] = (bf16)wq[i];
  if (i < CC * HID)     wob[i] = (bf16)wo[i];
}

// --------------------------------------------------- qkv GEMM + fused q-softmax
// grid (NN/128, 24, BB), block 256.  Output tile 64 rows x 128 cols.
// Double-buffered LDS staging: one barrier per 32-wide K-step.
__global__ __launch_bounds__(256) void k_qkv(const float* __restrict__ x,
                                             const bf16* __restrict__ wq,
                                             bf16* __restrict__ qkv) {
  __shared__ bf16 A2[2][64 * 40];
  __shared__ bf16 B2[2][128 * 40];
  __shared__ float outt[64 * 128];
  const int tid = threadIdx.x, lane = tid & 31, w = tid >> 5;
  const int ct = blockIdx.x * 128;
  const int rb = blockIdx.y;
  const int b = blockIdx.z;
  const int msub = w >> 1, ns0 = (w & 1) * 4;
  const int hi8 = (lane & 16) ? 8 : 0;
  v8f z = {0.f, 0.f, 0.f, 0.f, 0.f, 0.f, 0.f, 0.f};
  v8f acc[4] = {z, z, z, z};

  const int ar = tid >> 2, ak0 = (tid & 3) * 8;   // A staging: 8 bf16 per thread
  const int n4 = (tid & 31) * 4, kb = tid >> 5;   // B staging: 4x float4 per thread
  const float* xb = x + (size_t)b * CC * NN + ct;
  const bf16* wqb = wq + (size_t)rb * 64 * CC;

  auto stage = [&](bf16* Ab, bf16* Bb, int kt) {
    async_b128(&Ab[ar * 40 + ak0], wqb + (size_t)ar * CC + kt + ak0);
#pragma unroll
    for (int i = 0; i < 4; ++i) {
      int kk = kb + i * 8;
      v4f xv = *(const v4f*)&xb[(size_t)(kt + kk) * NN + n4];
#pragma unroll
      for (int j = 0; j < 4; ++j) Bb[(n4 + j) * 40 + kk] = (bf16)xv[j];
    }
  };

  stage(A2[0], B2[0], 0);
#pragma unroll
  for (int it = 0; it < 8; ++it) {
    wait_async();
    __syncthreads();
    if (it < 7) stage(A2[(it + 1) & 1], B2[(it + 1) & 1], (it + 1) * 32);
    const bf16* Ab = A2[it & 1];
    const bf16* Bb = B2[it & 1];
    v16bf a = ld_a(Ab, 40, msub * 16 + (lane & 15), 0, lane);
    v16bf b0 = ld_b(Bb, 40, (ns0 + 0) * 16 + (lane & 15), 0, lane);
    v16bf b1 = ld_b(Bb, 40, (ns0 + 1) * 16 + (lane & 15), 0, lane);
    v16bf b2 = ld_b(Bb, 40, (ns0 + 2) * 16 + (lane & 15), 0, lane);
    v16bf b3 = ld_b(Bb, 40, (ns0 + 3) * 16 + (lane & 15), 0, lane);
    acc[0] = wmma_bf16(a, b0, acc[0]);
    acc[1] = wmma_bf16(a, b1, acc[1]);
    acc[2] = wmma_bf16(a, b2, acc[2]);
    acc[3] = wmma_bf16(a, b3, acc[3]);
  }
  __syncthreads();
  // spill accumulators to LDS tile
#pragma unroll
  for (int j = 0; j < 4; ++j)
#pragma unroll
    for (int r = 0; r < 8; ++r)
      outt[(msub * 16 + r + hi8) * 128 + (ns0 + j) * 16 + (lane & 15)] = acc[j][r];
  __syncthreads();

  if (rb < 8) {  // q section: softmax over head-dim (the 64 rows), * D^-0.5
    if (tid < 128) {
      float mx = -1e30f;
      for (int r = 0; r < 64; ++r) mx = fmaxf(mx, outt[r * 128 + tid]);
      float s = 0.f;
      for (int r = 0; r < 64; ++r) {
        float e = __expf(outt[r * 128 + tid] - mx);
        outt[r * 128 + tid] = e;
        s += e;
      }
      float inv = 0.125f / s;  // SCALE = 64^-0.5
      for (int r = 0; r < 64; ++r) outt[r * 128 + tid] *= inv;
    }
    __syncthreads();
  }
  bf16* dst = qkv + ((size_t)b * (3 * HID) + rb * 64) * NN + ct;
#pragma unroll
  for (int i = 0; i < 32; ++i) {
    int lin = i * 256 + tid;
    int r = lin >> 7, c = lin & 127;
    dst[(size_t)r * NN + c] = (bf16)outt[r * 128 + c];
  }
}

// --------- context partials per (b,h,slice): flash-style k-softmax + sm(k)@v^T
// grid (HH, BB, 4), block 256.  64-wide K chunks, double-buffered.
__global__ __launch_bounds__(256) void k_ctx_part(const bf16* __restrict__ qkv,
                                                  float* __restrict__ ctxp,
                                                  float* __restrict__ maxb,
                                                  float* __restrict__ sumb) {
  __shared__ float rowmax[64];
  __shared__ bf16 A2[2][64 * 72];
  __shared__ bf16 B2[2][64 * 72];
  const int tid = threadIdx.x, lane = tid & 31, w = tid >> 5;
  const int h = blockIdx.x, b = blockIdx.y, s = blockIdx.z;
  const bf16* kbase = qkv + ((size_t)b * (3 * HID) + HID + h * 64) * NN + s * NSL;
  const bf16* vbase = qkv + ((size_t)b * (3 * HID) + 2 * HID + h * 64) * NN + s * NSL;
  const size_t vecbase = (size_t)((b * HH + h) * 4 + s) * 64;

  // phase 1: per-row max over this slice (wave w owns rows w, w+8, ..., w+56)
#pragma unroll
  for (int rr = 0; rr < 8; ++rr) {
    int d = rr * 8 + w;
    const bf16* kp = kbase + (size_t)d * NN;
    float mx = -1e30f;
    for (int n = lane; n < NSL; n += 32) mx = fmaxf(mx, (float)kp[n]);
#pragma unroll
    for (int o = 16; o; o >>= 1) mx = fmaxf(mx, __shfl_xor(mx, o, 32));
    if (lane == 0) {
      rowmax[d] = mx;
      maxb[vecbase + d] = mx;
    }
  }
  __syncthreads();

  const int msub = w >> 1, ns0 = (w & 1) * 2;
  const int drow = tid >> 2, kq = (tid & 3) * 16;  // thread owns row drow, 16 k's
  v8f z = {0.f, 0.f, 0.f, 0.f, 0.f, 0.f, 0.f, 0.f};
  v8f acc0 = z, acc1 = z;
  float psum = 0.f;
  const float rm = rowmax[drow];

  auto stage = [&](bf16* Ab, bf16* Bb, int n0) {
    const bf16* kp = kbase + (size_t)drow * NN + n0 + kq;
#pragma unroll
    for (int half = 0; half < 2; ++half) {
      v8bf kv = *(const v8bf*)(kp + half * 8);
      v8bf ev;
#pragma unroll
      for (int j = 0; j < 8; ++j) {
        float e = __expf((float)kv[j] - rm);
        psum += e;
        ev[j] = (bf16)e;
      }
      *(v8bf*)&Ab[drow * 72 + kq + half * 8] = ev;
    }
    async_b128(&Bb[drow * 72 + kq], vbase + (size_t)drow * NN + n0 + kq);
    async_b128(&Bb[drow * 72 + kq + 8], vbase + (size_t)drow * NN + n0 + kq + 8);
  };

  stage(A2[0], B2[0], 0);
#pragma unroll 2
  for (int it = 0; it < 16; ++it) {
    wait_async();
    __syncthreads();
    if (it < 15) stage(A2[(it + 1) & 1], B2[(it + 1) & 1], (it + 1) * 64);
    const bf16* Ab = A2[it & 1];
    const bf16* Bb = B2[it & 1];
#pragma unroll
    for (int kt = 0; kt < 64; kt += 32) {
      v16bf a = ld_a(Ab, 72, msub * 16 + (lane & 15), kt, lane);
      v16bf b0 = ld_b(Bb, 72, ns0 * 16 + (lane & 15), kt, lane);
      v16bf b1 = ld_b(Bb, 72, (ns0 + 1) * 16 + (lane & 15), kt, lane);
      acc0 = wmma_bf16(a, b0, acc0);
      acc1 = wmma_bf16(a, b1, acc1);
    }
  }
  // partial softmax denominators: 4 threads share a row
  float p2 = psum + __shfl_xor(psum, 1, 32);
  p2 += __shfl_xor(p2, 2, 32);
  if ((lane & 3) == 0) sumb[vecbase + drow] = p2;

  // write f32 partial context [d][e]
  const int hi8 = (lane & 16) ? 8 : 0;
  float* cdst = ctxp + vecbase * 64;
#pragma unroll
  for (int t = 0; t < 2; ++t) {
    v8f av = t ? acc1 : acc0;
    int e = (ns0 + t) * 16 + (lane & 15);
#pragma unroll
    for (int r = 0; r < 8; ++r) {
      int d = msub * 16 + r + hi8;
      cdst[(size_t)d * 64 + e] = av[r];
    }
  }
}

// ----------------- merge 4 context slices (rescale by exp(max_s - M)), -> bf16
// grid (BB*HH), block 256.
__global__ __launch_bounds__(256) void k_ctx_fin(const float* __restrict__ ctxp,
                                                 const float* __restrict__ maxb,
                                                 const float* __restrict__ sumb,
                                                 bf16* __restrict__ ctxT) {
  __shared__ float scl[4][64];
  __shared__ float rs[64];
  const int tid = threadIdx.x;
  const int bh = blockIdx.x;
  const float* mb = maxb + (size_t)bh * 4 * 64;
  const float* sb = sumb + (size_t)bh * 4 * 64;
  if (tid < 64) {
    int d = tid;
    float M = mb[d];
#pragma unroll
    for (int s = 1; s < 4; ++s) M = fmaxf(M, mb[s * 64 + d]);
    float rsum = 0.f;
#pragma unroll
    for (int s = 0; s < 4; ++s) {
      float e = __expf(mb[s * 64 + d] - M);
      scl[s][d] = e;
      rsum += sb[s * 64 + d] * e;
    }
    rs[d] = rsum;
  }
  __syncthreads();
  const float* cp = ctxp + (size_t)bh * 4 * 4096;
  bf16* cd = ctxT + (size_t)bh * 4096;
  int d = tid >> 2;
  float inv = 1.0f / rs[d];
#pragma unroll
  for (int j = 0; j < 16; ++j) {
    int e = (tid & 3) * 16 + j;
    float v = 0.f;
#pragma unroll
    for (int s = 0; s < 4; ++s) v += cp[(size_t)s * 4096 + d * 64 + e] * scl[s][d];
    cd[(size_t)e * 64 + d] = (bf16)(v * inv);  // store transposed: ctxT[e][d]
  }
}

// ------- fused: out = ctx^T@q per head, y = w_out@out + b, RMSNorm over C
// grid (NN/64, BB), block 256.  64-column tile.  3 barriers per head.
__global__ __launch_bounds__(256) void k_out(const bf16* __restrict__ qkv,
                                             const bf16* __restrict__ ctxT,
                                             const bf16* __restrict__ wo,
                                             const float* __restrict__ bout,
                                             const float* __restrict__ gnorm,
                                             float* __restrict__ y) {
  __shared__ bf16 Actx[64 * 72];
  __shared__ bf16 Bq[64 * 72];
  __shared__ bf16 outH[64 * 72];   // [n][e] -> B operand of second GEMM
  __shared__ bf16 Aw[256 * 72];    // w_out 256 x 64 head-slice
  __shared__ float blds[256];
  __shared__ float glds[256];
  __shared__ float colsq[64];
  const int tid = threadIdx.x, lane = tid & 31, w = tid >> 5;
  const int n0 = blockIdx.x * 64, b = blockIdx.y;
  const int hi8 = (lane & 16) ? 8 : 0;
  blds[tid] = bout[tid];
  glds[tid] = gnorm[tid];
  if (tid < 64) colsq[tid] = 0.f;
  v8f z = {0.f, 0.f, 0.f, 0.f, 0.f, 0.f, 0.f, 0.f};
  v8f acc2[2][4] = {{z, z, z, z}, {z, z, z, z}};

  for (int h = 0; h < HH; ++h) {
    __syncthreads();
    const bf16* csrc = ctxT + (size_t)(b * HH + h) * 64 * 64;
    const bf16* qsrc = qkv + ((size_t)b * (3 * HID) + h * 64) * NN + n0;
    {  // stage ctxT [e][d] via async DMA (2 x b128 per thread)
      int e = tid >> 2, d0 = (tid & 3) * 16;
      async_b128(&Actx[e * 72 + d0], csrc + (size_t)e * 64 + d0);
      async_b128(&Actx[e * 72 + d0 + 8], csrc + (size_t)e * 64 + d0 + 8);
    }
#pragma unroll
    for (int i = 0; i < 8; ++i) {  // stage full w_out head-slice via async DMA
      int c = tid + i * 256;
      int rr = c >> 3, kk0 = (c & 7) * 8;
      async_b128(&Aw[rr * 72 + kk0], wo + (size_t)rr * HID + h * 64 + kk0);
    }
#pragma unroll
    for (int i = 0; i < 2; ++i) {  // stage q transposed -> Bq[n][d]
      int c = tid + i * 256;
      int d = c >> 3, nb = (c & 7) * 8;
      v8bf qv = *(const v8bf*)(qsrc + (size_t)d * NN + nb);
#pragma unroll
      for (int j = 0; j < 8; ++j) Bq[(nb + j) * 72 + d] = qv[j];
    }
    wait_async();
    __syncthreads();
    // first GEMM: outT[e][n] = sum_d ctxT[e][d] * q[d][n]
    int msub = w >> 1, nsA = (w & 1) * 2;
    v8f t0 = z, t1 = z;
#pragma unroll
    for (int kt = 0; kt < 64; kt += 32) {
      v16bf a = ld_a(Actx, 72, msub * 16 + (lane & 15), kt, lane);
      v16bf q0 = ld_b(Bq, 72, nsA * 16 + (lane & 15), kt, lane);
      v16bf q1 = ld_b(Bq, 72, (nsA + 1) * 16 + (lane & 15), kt, lane);
      t0 = wmma_bf16(a, q0, t0);
      t1 = wmma_bf16(a, q1, t1);
    }
#pragma unroll
    for (int t = 0; t < 2; ++t) {  // outH[n][e] (bf16)
      v8f tv = t ? t1 : t0;
      int nn = (nsA + t) * 16 + (lane & 15);
#pragma unroll
      for (int r = 0; r < 8; ++r) outH[nn * 72 + msub * 16 + r + hi8] = (bf16)tv[r];
    }
    __syncthreads();
    // second GEMM (partial over this head): y += w_out[:, h*64:+64] @ outH
#pragma unroll
    for (int kt = 0; kt < 64; kt += 32) {
      v16bf bv0 = ld_b(outH, 72, 0 * 16 + (lane & 15), kt, lane);
      v16bf bv1 = ld_b(outH, 72, 1 * 16 + (lane & 15), kt, lane);
      v16bf bv2 = ld_b(outH, 72, 2 * 16 + (lane & 15), kt, lane);
      v16bf bv3 = ld_b(outH, 72, 3 * 16 + (lane & 15), kt, lane);
#pragma unroll
      for (int mi = 0; mi < 2; ++mi) {
        v16bf a = ld_a(Aw, 72, (w * 2 + mi) * 16 + (lane & 15), kt, lane);
        acc2[mi][0] = wmma_bf16(a, bv0, acc2[mi][0]);
        acc2[mi][1] = wmma_bf16(a, bv1, acc2[mi][1]);
        acc2[mi][2] = wmma_bf16(a, bv2, acc2[mi][2]);
        acc2[mi][3] = wmma_bf16(a, bv3, acc2[mi][3]);
      }
    }
  }
  __syncthreads();
  // epilogue: bias, column sum of squares (RMS over channels), scale, store
#pragma unroll
  for (int mi = 0; mi < 2; ++mi) {
    int msub = w * 2 + mi;
#pragma unroll
    for (int ns = 0; ns < 4; ++ns) {
      float p = 0.f;
#pragma unroll
      for (int r = 0; r < 8; ++r) {
        int row = msub * 16 + r + hi8;
        float v2 = acc2[mi][ns][r] + blds[row];
        acc2[mi][ns][r] = v2;
        p += v2 * v2;
      }
      atomicAdd(&colsq[ns * 16 + (lane & 15)], p);
    }
  }
  __syncthreads();
#pragma unroll
  for (int mi = 0; mi < 2; ++mi) {
    int msub = w * 2 + mi;
#pragma unroll
    for (int ns = 0; ns < 4; ++ns) {
      int col = ns * 16 + (lane & 15);
      float scl = 16.0f / fmaxf(sqrtf(colsq[col]), 1e-12f);  // sqrt(C)=16
#pragma unroll
      for (int r = 0; r < 8; ++r) {
        int row = msub * 16 + r + hi8;
        y[((size_t)b * CC + row) * NN + n0 + col] = acc2[mi][ns][r] * scl * glds[row];
      }
    }
  }
}

extern "C" void kernel_launch(void* const* d_in, const int* in_sizes, int n_in,
                              void* d_out, int out_size, void* d_ws, size_t ws_size,
                              hipStream_t stream) {
  (void)in_sizes; (void)n_in; (void)out_size; (void)ws_size;
  const float* x    = (const float*)d_in[0];
  const float* wqkv = (const float*)d_in[1];
  const float* wout = (const float*)d_in[2];
  const float* bo   = (const float*)d_in[3];
  const float* gn   = (const float*)d_in[4];
  float* y = (float*)d_out;

  char* ws = (char*)d_ws;
  size_t off = 0;
  auto take = [&](size_t bytes) -> char* {
    char* p = ws + off;
    off = (off + bytes + 255) & ~(size_t)255;
    return p;
  };
  bf16* wqkv_bf = (bf16*)take((size_t)3 * HID * CC * 2);
  bf16* wout_bf = (bf16*)take((size_t)CC * HID * 2);
  bf16* qkv_bf  = (bf16*)take((size_t)BB * 3 * HID * NN * 2);
  bf16* ctxT    = (bf16*)take((size_t)BB * HH * 64 * 64 * 2);
  float* ctxp   = (float*)take((size_t)BB * HH * 4 * 64 * 64 * 4);
  float* maxb   = (float*)take((size_t)BB * HH * 4 * 64 * 4);
  float* sumb   = (float*)take((size_t)BB * HH * 4 * 64 * 4);

  k_prep<<<(3 * HID * CC + 255) / 256, 256, 0, stream>>>(wqkv, wout, wqkv_bf, wout_bf);
  k_qkv<<<dim3(NN / 128, 24, BB), 256, 0, stream>>>(x, wqkv_bf, qkv_bf);
  k_ctx_part<<<dim3(HH, BB, 4), 256, 0, stream>>>(qkv_bf, ctxp, maxb, sumb);
  k_ctx_fin<<<dim3(BB * HH), 256, 0, stream>>>(ctxp, maxb, sumb, ctxT);
  k_out<<<dim3(NN / 64, BB), 256, 0, stream>>>(qkv_bf, ctxT, wout_bf, bo, gn, y);
}
